// WindowAttention_4337916970083
// MI455X (gfx1250) — matrix operand order repaired and checked by reading, not compiled, not run on hardware
//
#include <hip/hip_runtime.h>
#include <hip/hip_bf16.h>
#include <math.h>

// ---------------------------------------------------------------------------
// Swin-V2 window attention, fused per-window, gfx1250 WMMA (wave32).
// B_=4096 windows, N=64 tokens, C=192, H=6 heads, D=32, nW=64 mask windows.
// ---------------------------------------------------------------------------

typedef __attribute__((ext_vector_type(16))) _Float16 v16h;
typedef __attribute__((ext_vector_type(8)))  _Float16 v8h;
typedef __attribute__((ext_vector_type(8)))  float    v8f;

#define DIMC 192
#define NTOK 64
#define NHEAD 6
#define HDIM 32
#define NWIN 64

// LDS strides (in elements). All multiples of 8 halves -> 16B-aligned b128 loads.
#define SX   200   // x / O rows (f16), 64 rows
#define SQK  40    // q,k rows (f16), 64 rows
#define SVT  72    // v-transposed rows (f16), 32 rows (d-major)
#define SPH  72    // softmax probs f16, 64 rows
#define SPF  66    // scores f32, 64 rows

#define SMEM_BYTES ((64*SX + 64*SX + 64*SQK + 64*SQK + 32*SVT + 64*SPH)*2 + 64*SPF*4)

// Workspace offsets (bytes)
#define WS_QKVW16  0                         // 576*192 f16 = 221184
#define WS_PROJW16 221184                    // 192*192 f16 = 73728
#define WS_QKVB    294912                    // 576 f32
#define WS_SCALE   297216                    // 6 f32
#define WS_RPB     297472                    // 225*6 f32
#define WS_BIAS16  303104                    // 6*64*64 f32 = 98304

union F16x16 { v16h v; v8h h[2]; };

// A fragment: 16x32 (MxK), lane m=l&15, half=l>>4.
// elems 0..7 -> K = k0 + half*8 + i ; elems 8..15 -> K = k0 + 16 + half*8 + i
__device__ __forceinline__ v16h ldA(const _Float16* base, int stride, int m0, int k0, int lane) {
  const int m = m0 + (lane & 15);
  const int hf = lane >> 4;
  const _Float16* p = base + m * stride + k0 + hf * 8;
  F16x16 f;
  f.h[0] = *(const v8h*)(p);
  f.h[1] = *(const v8h*)(p + 16);
  return f.v;
}

// B fragment: 32x16 (KxN), B[k][n] = W[n0+n][k] (W row-major, stride "stride").
// lane n=l&15, half=l>>4; elems 0..7 -> K = k0 + half*8 + i ; elems 8..15 -> K = k0 + 16 + half*8 + i
__device__ __forceinline__ v16h ldB(const _Float16* base, int stride, int n0, int k0, int lane) {
  const int n = n0 + (lane & 15);
  const int hf = lane >> 4;
  const _Float16* p = base + n * stride + k0 + hf * 8;
  F16x16 f;
  f.h[0] = *(const v8h*)(p);
  f.h[1] = *(const v8h*)(p + 16);
  return f.v;
}

__device__ __forceinline__ v8f vzero8() {
  v8f z = {0.f,0.f,0.f,0.f,0.f,0.f,0.f,0.f};
  return z;
}

// ---------------------------------------------------------------------------
// Prep 0: f32->f16 weights, concat qkv bias, clipped-exp logit scale
// ---------------------------------------------------------------------------
__global__ void prep_weights(const float* __restrict__ qkv_w,
                             const float* __restrict__ proj_w,
                             const float* __restrict__ q_bias,
                             const float* __restrict__ v_bias,
                             const float* __restrict__ logit_scale,
                             _Float16* __restrict__ qkvw16,
                             _Float16* __restrict__ projw16,
                             float* __restrict__ qkvb,
                             float* __restrict__ scale6) {
  int idx = blockIdx.x * blockDim.x + threadIdx.x;
  if (idx < 576 * 192) qkvw16[idx] = (_Float16)qkv_w[idx];
  if (idx < 192 * 192) projw16[idx] = (_Float16)proj_w[idx];
  if (idx < 576) {
    int s = idx / 192, r = idx % 192;
    qkvb[idx] = (s == 0) ? q_bias[r] : ((s == 2) ? v_bias[r] : 0.0f);
  }
  if (idx < 6) scale6[idx] = __expf(fminf(logit_scale[idx], 4.60517019f)); // ln(100)
}

// ---------------------------------------------------------------------------
// Prep 1: continuous-position-bias MLP -> rpb_table[225][6]
// ---------------------------------------------------------------------------
__global__ void prep_cpb(const float* __restrict__ w1,   // [512][2]
                         const float* __restrict__ b1,   // [512]
                         const float* __restrict__ w2,   // [6][512]
                         float* __restrict__ rpb) {      // [225][6]
  int t = threadIdx.x;
  if (t >= 225) return;
  int di = t / 15, dj = t % 15;
  float c0 = (float)(di - 7) * (8.0f / 7.0f);
  float c1 = (float)(dj - 7) * (8.0f / 7.0f);
  c0 = ((c0 >= 0.f) ? 1.f : -1.f) * log2f(fabsf(c0) + 1.0f) * (1.0f / 3.0f);
  c1 = ((c1 >= 0.f) ? 1.f : -1.f) * log2f(fabsf(c1) + 1.0f) * (1.0f / 3.0f);
  float acc[6] = {0.f, 0.f, 0.f, 0.f, 0.f, 0.f};
  for (int j = 0; j < 512; ++j) {
    float hm = fmaxf(w1[2 * j] * c0 + w1[2 * j + 1] * c1 + b1[j], 0.f);
#pragma unroll
    for (int h = 0; h < 6; ++h) acc[h] += hm * w2[h * 512 + j];
  }
#pragma unroll
  for (int h = 0; h < 6; ++h) rpb[t * 6 + h] = acc[h];
}

// ---------------------------------------------------------------------------
// Prep 2: gather via rel-index + 16*sigmoid -> bias16[6][64][64]
// ---------------------------------------------------------------------------
__global__ void prep_bias16(const float* __restrict__ rpb, float* __restrict__ bias16) {
  int g = blockIdx.x * blockDim.x + threadIdx.x;
  if (g >= 6 * 64 * 64) return;
  int h = g / 4096, r = g % 4096, n = r / 64, m = r % 64;
  int dy = (n >> 3) - (m >> 3) + 7;
  int dx = (n & 7) - (m & 7) + 7;
  float v = rpb[(dy * 15 + dx) * 6 + h];
  bias16[g] = 16.0f / (1.0f + __expf(-v));
}

// ---------------------------------------------------------------------------
// Main fused kernel: one workgroup (8 wave32) per window.
// ---------------------------------------------------------------------------
__global__ __launch_bounds__(256)
void win_attn(const float* __restrict__ x,
              const float* __restrict__ mask,
              const _Float16* __restrict__ qkvw16,
              const _Float16* __restrict__ projw16,
              const float* __restrict__ qkvb,
              const float* __restrict__ scale6,
              const float* __restrict__ bias16,
              const float* __restrict__ proj_b,
              float* __restrict__ out) {
  extern __shared__ __attribute__((aligned(16))) char smem[];
  _Float16* sX  = (_Float16*)smem;           // [64][SX]  staged x (f16)
  _Float16* sO  = sX + 64 * SX;              // [64][SX]  output accumulator (f16)
  _Float16* sQ  = sO + 64 * SX;              // [64][SQK] q (normalized*scale)
  _Float16* sK  = sQ + 64 * SQK;             // [64][SQK] k (normalized)
  _Float16* sVt = sK + 64 * SQK;             // [32][SVT] v transposed (d-major)
  _Float16* sPh = sVt + 32 * SVT;            // [64][SPH] softmax probs f16
  float*    sP  = (float*)(sPh + 64 * SPH);  // [64][SPF] scores f32

  const int tid  = threadIdx.x;
  const int lane = tid & 31;
  const int wv   = tid >> 5;
  const int b    = blockIdx.x;
  const int w    = b & (NWIN - 1);
  const int nn   = lane & 15;
  const int hf   = lane >> 4;

  // Stage x[b] (64x192 f32) into LDS as f16
  {
    const float4* xg = (const float4*)(x + (size_t)b * (NTOK * DIMC));
    for (int i = tid; i < NTOK * DIMC / 4; i += 256) {
      float4 v = xg[i];
      int row = i / (DIMC / 4);
      int c = (i % (DIMC / 4)) * 4;
      _Float16* d = sX + row * SX + c;
      d[0] = (_Float16)v.x; d[1] = (_Float16)v.y;
      d[2] = (_Float16)v.z; d[3] = (_Float16)v.w;
    }
  }
  __syncthreads();

  for (int h = 0; h < NHEAD; ++h) {
    const float scale_h = scale6[h];

    // ---- Phase 1: q/k/v slices for head h: [64x32] each, K=192 -------------
    // 24 tiles (3 mats x 2 col-halves x 4 row-tiles), 3 per wave.
    for (int tt = wv; tt < 24; tt += 8) {
      int s  = tt >> 3;          // 0=q 1=k 2=v
      int t  = (tt >> 2) & 1;    // 16-col half
      int mt = tt & 3;           // 16-row tile
      int j0 = s * DIMC + h * HDIM + t * 16;
      int m0 = mt * 16;
      v8f acc = vzero8();
#pragma unroll
      for (int kk = 0; kk < 6; ++kk) {
        v16h a  = ldA(sX, SX, m0, kk * 32, lane);
        v16h bf = ldB(qkvw16, DIMC, j0, kk * 32, lane);
        acc = __builtin_amdgcn_wmma_f32_16x16x32_f16(false, a, false, bf,
                                                     (short)0, acc, false, false);
      }
      float bias = qkvb[j0 + nn];
      if (s == 2) {
        // store V transposed: sVt[d][token]
#pragma unroll
        for (int r = 0; r < 8; ++r)
          sVt[(t * 16 + nn) * SVT + m0 + r + hf * 8] = (_Float16)(acc[r] + bias);
      } else {
        _Float16* dst = (s == 0) ? sQ : sK;
#pragma unroll
        for (int r = 0; r < 8; ++r)
          dst[(m0 + r + hf * 8) * SQK + t * 16 + nn] = (_Float16)(acc[r] + bias);
      }
    }
    __syncthreads();

    // ---- Phase 1.5: cosine-normalize q,k rows; fold logit scale into q -----
    if (tid < 128) {
      int row = tid & 63;
      int which = tid >> 6;                  // 0=q 1=k
      _Float16* buf = which ? sK : sQ;
      float ss = 0.f;
#pragma unroll
      for (int d = 0; d < HDIM; ++d) { float v = (float)buf[row * SQK + d]; ss += v * v; }
      float inv = 1.0f / sqrtf(fmaxf(ss, 1e-24f));
      if (!which) inv *= scale_h;
#pragma unroll
      for (int d = 0; d < HDIM; ++d)
        buf[row * SQK + d] = (_Float16)((float)buf[row * SQK + d] * inv);
    }
    __syncthreads();

    // ---- Phase 2a: S = (qn*scale) @ kn^T + cpb + mask ----------------------
    {
      const float* bb = bias16 + h * (NTOK * NTOK);
      const float* mm = mask + (size_t)w * (NTOK * NTOK);
      for (int tt = wv; tt < 16; tt += 8) {
        int m0 = (tt >> 2) * 16, n0 = (tt & 3) * 16;
        v16h a  = ldA(sQ, SQK, m0, 0, lane);
        v16h bf = ldB(sK, SQK, n0, 0, lane);
        v8f acc = vzero8();
        acc = __builtin_amdgcn_wmma_f32_16x16x32_f16(false, a, false, bf,
                                                     (short)0, acc, false, false);
#pragma unroll
        for (int r = 0; r < 8; ++r) {
          int row = m0 + r + hf * 8, col = n0 + nn;
          sP[row * SPF + col] = acc[r] + bb[row * NTOK + col] + mm[row * NTOK + col];
        }
      }
    }
    __syncthreads();

    // ---- Phase 2.5: row softmax -> f16 probs -------------------------------
    if (tid < NTOK) {
      float* p = sP + tid * SPF;
      float mx = -3.4e38f;
      for (int j = 0; j < NTOK; ++j) mx = fmaxf(mx, p[j]);
      float sum = 0.f;
      for (int j = 0; j < NTOK; ++j) { float e = __expf(p[j] - mx); p[j] = e; sum += e; }
      float inv = 1.0f / sum;
      _Float16* ph = sPh + tid * SPH;
      for (int j = 0; j < NTOK; ++j) ph[j] = (_Float16)(p[j] * inv);
    }
    __syncthreads();

    // ---- Phase 2b: O_h = P @ V (8 tiles, 1 per wave) -----------------------
    {
      int m0 = (wv >> 1) * 16, n0 = (wv & 1) * 16;
      v8f acc = vzero8();
#pragma unroll
      for (int kk = 0; kk < 2; ++kk) {
        v16h a  = ldA(sPh, SPH, m0, kk * 32, lane);
        v16h bf = ldB(sVt, SVT, n0, kk * 32, lane);   // B[k][d] = Vt[d][k]
        acc = __builtin_amdgcn_wmma_f32_16x16x32_f16(false, a, false, bf,
                                                     (short)0, acc, false, false);
      }
#pragma unroll
      for (int r = 0; r < 8; ++r)
        sO[(m0 + r + hf * 8) * SX + h * HDIM + n0 + nn] = (_Float16)acc[r];
    }
    __syncthreads();
  }

  // ---- Phase 3: out = O @ proj_w^T + proj_b (48 tiles, 6 per wave) ---------
  {
    float* og = out + (size_t)b * (NTOK * DIMC);
    for (int tt = wv; tt < 48; tt += 8) {
      int m0 = (tt / 12) * 16, c0 = (tt % 12) * 16;
      v8f acc = vzero8();
#pragma unroll
      for (int kk = 0; kk < 6; ++kk) {
        v16h a  = ldA(sO, SX, m0, kk * 32, lane);
        v16h bf = ldB(projw16, DIMC, c0, kk * 32, lane);
        acc = __builtin_amdgcn_wmma_f32_16x16x32_f16(false, a, false, bf,
                                                     (short)0, acc, false, false);
      }
      float pb = proj_b[c0 + nn];
#pragma unroll
      for (int r = 0; r < 8; ++r)
        og[(m0 + r + hf * 8) * DIMC + c0 + nn] = acc[r] + pb;
    }
  }
}

// ---------------------------------------------------------------------------
extern "C" void kernel_launch(void* const* d_in, const int* in_sizes, int n_in,
                              void* d_out, int out_size, void* d_ws, size_t ws_size,
                              hipStream_t stream) {
  (void)in_sizes; (void)n_in; (void)out_size; (void)ws_size;
  const float* x           = (const float*)d_in[0];
  const float* mask        = (const float*)d_in[1];
  const float* qkv_w       = (const float*)d_in[2];
  const float* q_bias      = (const float*)d_in[3];
  const float* v_bias      = (const float*)d_in[4];
  const float* logit_scale = (const float*)d_in[5];
  const float* cpb_w1      = (const float*)d_in[6];
  const float* cpb_b1      = (const float*)d_in[7];
  const float* cpb_w2      = (const float*)d_in[8];
  const float* proj_w      = (const float*)d_in[9];
  const float* proj_b      = (const float*)d_in[10];
  float* out = (float*)d_out;

  char* ws = (char*)d_ws;
  _Float16* qkvw16  = (_Float16*)(ws + WS_QKVW16);
  _Float16* projw16 = (_Float16*)(ws + WS_PROJW16);
  float*    qkvb    = (float*)(ws + WS_QKVB);
  float*    scale6  = (float*)(ws + WS_SCALE);
  float*    rpb     = (float*)(ws + WS_RPB);
  float*    bias16  = (float*)(ws + WS_BIAS16);

  prep_weights<<<576, 256, 0, stream>>>(qkv_w, proj_w, q_bias, v_bias, logit_scale,
                                        qkvw16, projw16, qkvb, scale6);
  prep_cpb<<<1, 256, 0, stream>>>(cpb_w1, cpb_b1, cpb_w2, rpb);
  prep_bias16<<<96, 256, 0, stream>>>(rpb, bias16);

  hipFuncSetAttribute(reinterpret_cast<const void*>(win_attn),
                      hipFuncAttributeMaxDynamicSharedMemorySize, SMEM_BYTES);
  win_attn<<<4096, 256, SMEM_BYTES, stream>>>(x, mask, qkvw16, projw16, qkvb,
                                              scale6, bias16, proj_b, out);
}
